// MoE_86629490361052
// MI455X (gfx1250) — compile-verified, matrix operand order checked by reading, not run on hardware
//
#include <hip/hip_runtime.h>
#include <hip/hip_bf16.h>

// Problem constants (from reference setup_inputs)
#define S_TOK 8192   // B*T
#define MDIM  1024
#define EEXP  8
#define FDIM  4096
#define CCAP  1024   // capacity = S/E

typedef __attribute__((ext_vector_type(16))) __bf16          v16bf;
typedef __attribute__((ext_vector_type(8)))  float           v8f;
typedef __attribute__((ext_vector_type(4)))  unsigned int    v4u;
typedef __attribute__((ext_vector_type(2)))  unsigned int    v2u;
typedef __attribute__((ext_vector_type(4)))  float           v4f;
typedef __attribute__((ext_vector_type(4)))  int             v4i;

// Address-space qualified element types for the async global->LDS builtin
typedef __attribute__((address_space(1))) v4i g_v4i;   // global
typedef __attribute__((address_space(3))) v4i s_v4i;   // LDS

#if __has_builtin(__builtin_amdgcn_global_load_async_to_lds_b128)
#define HAVE_ASYNC_LDS 1
#else
#define HAVE_ASYNC_LDS 0
#endif

union FragBF {           // 32 bytes: one WMMA bf16 A/B fragment per lane
    v4u   u[2];
    v16bf v;
};

// f32 -> bf16, round-half-up: one v_add; the >>16 folds into d16_hi stores.
__device__ __forceinline__ unsigned int f2bf_hi(float f) {
    return __builtin_bit_cast(unsigned int, f) + 0x8000u;  // bf16 in [31:16]
}
__device__ __forceinline__ unsigned short f2bf(float f) {
    return (unsigned short)(f2bf_hi(f) >> 16);
}

// Packed f32x2 -> bf16x2 (single instruction when the builtin exists)
__device__ __forceinline__ unsigned int f2bf_pk(float lo, float hi) {
#if __has_builtin(__builtin_amdgcn_cvt_pk_bf16_f32)
    auto r = __builtin_amdgcn_cvt_pk_bf16_f32(lo, hi);
    static_assert(sizeof(r) == 4, "cvt_pk_bf16 size");
    return __builtin_bit_cast(unsigned int, r);
#else
    return (f2bf_hi(lo) >> 16) | (f2bf_hi(hi) & 0xFFFF0000u);
#endif
}

// Branchless tanh-GELU: 0.5*v*(1+tanh(c*(v+0.044715 v^3))) == v*sigmoid(2u)
__device__ __forceinline__ float gelu_fast(float v) {
    float u = 0.7978845608028654f * (v + 0.044715f * v * v * v);
    return v * __builtin_amdgcn_rcpf(1.0f + __expf(-2.0f * u));
}

// ---------------------------------------------------------------------------
// 1) Gating: one wave32 per token. logits = x[s]·wg, softmax, top-1.
// ---------------------------------------------------------------------------
__global__ __launch_bounds__(256) void gate_kernel(
    const float* __restrict__ x, const float* __restrict__ wg,
    int* __restrict__ idx, float* __restrict__ prob)
{
    const int lane = threadIdx.x & 31;
    const int s    = blockIdx.x * 8 + (threadIdx.x >> 5);

    float acc[EEXP];
    #pragma unroll
    for (int e = 0; e < EEXP; ++e) acc[e] = 0.0f;

    const float* xs = x + (size_t)s * MDIM;
    for (int m = lane; m < MDIM; m += 32) {
        float xv = xs[m];
        v4f w0 = *(const v4f*)(wg + (size_t)m * EEXP);
        v4f w1 = *(const v4f*)(wg + (size_t)m * EEXP + 4);
        acc[0] += xv * w0.x; acc[1] += xv * w0.y;
        acc[2] += xv * w0.z; acc[3] += xv * w0.w;
        acc[4] += xv * w1.x; acc[5] += xv * w1.y;
        acc[6] += xv * w1.z; acc[7] += xv * w1.w;
    }
    #pragma unroll
    for (int off = 16; off > 0; off >>= 1)
        #pragma unroll
        for (int e = 0; e < EEXP; ++e) acc[e] += __shfl_xor(acc[e], off, 32);

    if (lane == 0) {
        float mx = acc[0]; int am = 0;
        #pragma unroll
        for (int e = 1; e < EEXP; ++e) if (acc[e] > mx) { mx = acc[e]; am = e; }
        float sum = 0.0f;
        #pragma unroll
        for (int e = 0; e < EEXP; ++e) sum += __expf(acc[e] - mx);
        idx[s]  = am;
        prob[s] = 1.0f / sum;   // softmax prob of the argmax expert
    }
}

// ---------------------------------------------------------------------------
// 2) Deterministic token-order per-expert cumsum + capacity drop.
// ---------------------------------------------------------------------------
__global__ __launch_bounds__(1024) void scan_kernel(
    const int* __restrict__ idx, int* __restrict__ loc)
{
    __shared__ int sc[EEXP][1024];
    const int t = threadIdx.x;
    const int base = t * 8;

    int cnt[EEXP];
    #pragma unroll
    for (int e = 0; e < EEXP; ++e) cnt[e] = 0;
    int myidx[8];
    #pragma unroll
    for (int j = 0; j < 8; ++j) { myidx[j] = idx[base + j]; cnt[myidx[j]]++; }

    #pragma unroll
    for (int e = 0; e < EEXP; ++e) sc[e][t] = cnt[e];
    __syncthreads();

    for (int off = 1; off < 1024; off <<= 1) {
        int v[EEXP];
        #pragma unroll
        for (int e = 0; e < EEXP; ++e) v[e] = (t >= off) ? sc[e][t - off] : 0;
        __syncthreads();
        #pragma unroll
        for (int e = 0; e < EEXP; ++e) sc[e][t] += v[e];
        __syncthreads();
    }

    int pos[EEXP];
    #pragma unroll
    for (int e = 0; e < EEXP; ++e) pos[e] = sc[e][t] - cnt[e];  // exclusive prefix

    #pragma unroll
    for (int j = 0; j < 8; ++j) {
        int e = myidx[j];
        int p = pos[e]++;
        loc[base + j] = (p < CCAP) ? p : -1;   // drop overflow tokens
    }
}

// ---------------------------------------------------------------------------
// 3) Dispatch: gather kept tokens into bf16 [E][C][M].
// ---------------------------------------------------------------------------
__global__ __launch_bounds__(256) void dispatch_kernel(
    const float* __restrict__ x, const int* __restrict__ idx,
    const int* __restrict__ loc, unsigned short* __restrict__ disp)
{
    const int s = blockIdx.x;
    const int l = loc[s];
    if (l < 0) return;
    const int e = idx[s];
    const v4f* src = (const v4f*)(x + (size_t)s * MDIM);
    unsigned short* dst = disp + ((size_t)e * CCAP + l) * MDIM;
    const int i = threadIdx.x;                 // 256 threads * 4 elems = 1024
    v4f v = src[i];
    v2u o;
    o.x = f2bf_pk(v.x, v.y);
    o.y = f2bf_pk(v.z, v.w);
    *(v2u*)(dst + i * 4) = o;
}

// ---------------------------------------------------------------------------
// 4/5) Grouped GEMM on WMMA bf16: Out[e] = act(A[e] @ Bw[e] + bias[e])
//   A: [E][ROWS][K] bf16 row-major -> async global->LDS DMA (ASYNCcnt).
//   Bw: [E][K][N] fp32 -> bf16 (1 v_add + d16_hi store) during the LDS fill;
//   weights are read from HBM exactly once, as fp32.
//   128x128 tile / 256-thread WG, BK=32, double-buffered LDS, one barrier
//   per k-step, next tile prefetched under the WMMAs.
// ---------------------------------------------------------------------------
template<bool GELU_BF16>
__global__ __launch_bounds__(256) void moe_gemm_kernel(
    const unsigned short* __restrict__ Aall,
    const float*          __restrict__ Ball,
    const float*          __restrict__ biasAll,
    unsigned short*       __restrict__ OutBf,
    float*                __restrict__ OutF,
    int ROWS, int K, int N)
{
    // pitch 40 shorts (80B): keeps 16B alignment, makes the b128 fragment
    // loads bank-conflict free (20*r mod 64 distinct for r in [0,16)).
    __shared__ unsigned short lds_a[2][128 * 40];
    __shared__ unsigned short lds_b[2][128 * 40];

    const int e = blockIdx.z;
    const unsigned short* A  = Aall    + (size_t)e * ROWS * K;
    const float*          Bw = Ball    + (size_t)e * K * N;
    const float*          bi = biasAll + (size_t)e * N;

    const int tid  = threadIdx.x;
    const int lane = tid & 31;
    const int wave = tid >> 5;
    const int wm   = wave >> 2;          // 0..1 : 64-row stripe
    const int wn   = wave & 3;           // 0..3 : 32-col stripe
    const int rowBase = blockIdx.y * 128;
    const int colBase = blockIdx.x * 128;
    const int khalf = lane >> 4;         // ISA bf16 A/B lane layout
    const int lrow  = lane & 15;

    // Per-thread staging coordinates (loop-invariant)
    const int a_row = tid >> 2;          // 0..63  (+64 for second chunk)
    const int a_kg  = tid & 3;           // 8-elem k-group
    const int b_k0  = tid >> 5;          // 0..7   (+8*it)
    const int b_n4  = (tid & 31) * 4;    // 0..124

    v8f acc[4][2];
    #pragma unroll
    for (int i = 0; i < 4; ++i)
        #pragma unroll
        for (int j = 0; j < 2; ++j)
            #pragma unroll
            for (int k = 0; k < 8; ++k) acc[i][j][k] = 0.0f;

#if !HAVE_ASYNC_LDS
    v4u areg[2];
#endif
    v4f breg[4];

    // A tile: either async DMA straight into LDS (ASYNCcnt) or reg staging.
    auto loadA = [&](int kk, int buf) {
        #pragma unroll
        for (int it = 0; it < 2; ++it) {
            const unsigned short* gp =
                A + (size_t)(rowBase + a_row + it * 64) * K + kk + a_kg * 8;
            unsigned short* lp = &lds_a[buf][(a_row + it * 64) * 40 + a_kg * 8];
#if HAVE_ASYNC_LDS
            __builtin_amdgcn_global_load_async_to_lds_b128(
                (g_v4i*)gp, (s_v4i*)lp, 0, 0);
#else
            (void)lp;
            areg[it] = *(const v4u*)gp;
#endif
        }
    };
    auto drainA = [&](int buf) {
#if !HAVE_ASYNC_LDS
        #pragma unroll
        for (int it = 0; it < 2; ++it)
            *(v4u*)&lds_a[buf][(a_row + it * 64) * 40 + a_kg * 8] = areg[it];
#else
        (void)buf;
#endif
    };
    auto waitA = [&]() {
#if HAVE_ASYNC_LDS
#if __has_builtin(__builtin_amdgcn_s_wait_asynccnt)
        __builtin_amdgcn_s_wait_asynccnt(0);
#else
        asm volatile("s_wait_asynccnt 0" ::: "memory");
#endif
#endif
    };

    auto gloadB = [&](int kk) {
        #pragma unroll
        for (int it = 0; it < 4; ++it)
            breg[it] = *(const v4f*)(Bw + (size_t)(kk + b_k0 + it * 8) * N
                                        + colBase + b_n4);
    };
    // Convert fp32->bf16 (1 v_add each) and store high halves (d16_hi).
    auto lstoreB = [&](int buf) {
        #pragma unroll
        for (int it = 0; it < 4; ++it) {
            int k = b_k0 + it * 8;
            lds_b[buf][(b_n4 + 0) * 40 + k] = (unsigned short)(f2bf_hi(breg[it].x) >> 16);
            lds_b[buf][(b_n4 + 1) * 40 + k] = (unsigned short)(f2bf_hi(breg[it].y) >> 16);
            lds_b[buf][(b_n4 + 2) * 40 + k] = (unsigned short)(f2bf_hi(breg[it].z) >> 16);
            lds_b[buf][(b_n4 + 3) * 40 + k] = (unsigned short)(f2bf_hi(breg[it].w) >> 16);
        }
    };

    // Prologue: tile 0 into buffer 0.
    loadA(0, 0);
    gloadB(0);
    drainA(0);
    lstoreB(0);
    waitA();
    int buf = 0;

    for (int kk = 0; kk < K; kk += 32) {
        __syncthreads();                         // lds[buf] ready for all waves
        const bool more = (kk + 32) < K;
        if (more) { loadA(kk + 32, buf ^ 1); gloadB(kk + 32); }

        // Fragment loads per ISA 16-bit layout:
        // lanes 0-15: K 0..7 then 16..23 ; lanes 16-31: K 8..15 then 24..31
        FragBF af[4], bfv[2];
        #pragma unroll
        for (int mt = 0; mt < 4; ++mt) {
            int r = wm * 64 + mt * 16 + lrow;
            af[mt].u[0] = *(const v4u*)&lds_a[buf][r * 40 + khalf * 8];
            af[mt].u[1] = *(const v4u*)&lds_a[buf][r * 40 + 16 + khalf * 8];
        }
        #pragma unroll
        for (int nt = 0; nt < 2; ++nt) {
            int n = wn * 32 + nt * 16 + lrow;
            bfv[nt].u[0] = *(const v4u*)&lds_b[buf][n * 40 + khalf * 8];
            bfv[nt].u[1] = *(const v4u*)&lds_b[buf][n * 40 + 16 + khalf * 8];
        }
        #pragma unroll
        for (int mt = 0; mt < 4; ++mt)
            #pragma unroll
            for (int nt = 0; nt < 2; ++nt)
                acc[mt][nt] = __builtin_amdgcn_wmma_f32_16x16x32_bf16(
                    false, af[mt].v, false, bfv[nt].v,
                    (short)0, acc[mt][nt], false, false);

        if (more) { drainA(buf ^ 1); lstoreB(buf ^ 1); waitA(); }
        buf ^= 1;
    }

    // Epilogue. C/D layout: lanes 0-15 hold M=j, lanes 16-31 hold M=8+j.
    float bnt[2];
    #pragma unroll
    for (int nt = 0; nt < 2; ++nt)
        bnt[nt] = bi[colBase + wn * 32 + nt * 16 + lrow];

    #pragma unroll
    for (int mt = 0; mt < 4; ++mt) {
        #pragma unroll
        for (int nt = 0; nt < 2; ++nt) {
            const int col = colBase + wn * 32 + nt * 16 + lrow;
            const size_t obase = (size_t)e * ROWS * N + col;
            #pragma unroll
            for (int j = 0; j < 8; j += 2) {
                int row = rowBase + wm * 64 + mt * 16 + khalf * 8 + j;
                float v0 = acc[mt][nt][j + 0] + bnt[nt];
                float v1 = acc[mt][nt][j + 1] + bnt[nt];
                if constexpr (GELU_BF16) {
                    unsigned int r0 = f2bf_hi(gelu_fast(v0));
                    unsigned int r1 = f2bf_hi(gelu_fast(v1));
                    OutBf[obase + (size_t)(row + 0) * N] = (unsigned short)(r0 >> 16);
                    OutBf[obase + (size_t)(row + 1) * N] = (unsigned short)(r1 >> 16);
                } else {
                    OutF[obase + (size_t)(row + 0) * N] = v0;
                    OutF[obase + (size_t)(row + 1) * N] = v1;
                }
            }
        }
    }
}

// ---------------------------------------------------------------------------
// 6) Combine: out[s] = prob[s] * expert_out[e, loc]  (zeros for dropped).
// ---------------------------------------------------------------------------
__global__ __launch_bounds__(256) void combine_kernel(
    const float* __restrict__ eout, const int* __restrict__ idx,
    const int* __restrict__ loc, const float* __restrict__ prob,
    float* __restrict__ out)
{
    const int s = blockIdx.x;
    const int l = loc[s];
    const int i = threadIdx.x;
    v4f* dst = (v4f*)(out + (size_t)s * MDIM);
    if (l < 0) {
        v4f z; z.x = 0.f; z.y = 0.f; z.z = 0.f; z.w = 0.f;
        dst[i] = z;
        return;
    }
    const int e = idx[s];
    const float g = prob[s];
    v4f v = ((const v4f*)(eout + ((size_t)e * CCAP + l) * MDIM))[i];
    v.x *= g; v.y *= g; v.z *= g; v.w *= g;
    dst[i] = v;
}

// ---------------------------------------------------------------------------
extern "C" void kernel_launch(void* const* d_in, const int* in_sizes, int n_in,
                              void* d_out, int out_size, void* d_ws, size_t ws_size,
                              hipStream_t stream)
{
    const float* x  = (const float*)d_in[0];   // [S, M]
    const float* wg = (const float*)d_in[1];   // [M, E]
    const float* w1 = (const float*)d_in[2];   // [E, M, F]
    const float* b1 = (const float*)d_in[3];   // [E, F]
    const float* w2 = (const float*)d_in[4];   // [E, F, M]
    const float* b2 = (const float*)d_in[5];   // [E, M]
    float* out = (float*)d_out;

    char*  ws  = (char*)d_ws;
    size_t off = 0;
    auto take = [&](size_t bytes) -> char* {
        char* p = ws + off;
        off = (off + bytes + 255) & ~(size_t)255;
        return p;
    };
    int*            idx  = (int*)           take((size_t)S_TOK * 4);
    float*          prob = (float*)         take((size_t)S_TOK * 4);
    int*            loc  = (int*)           take((size_t)S_TOK * 4);
    unsigned short* disp = (unsigned short*)take((size_t)EEXP * CCAP * MDIM * 2);
    unsigned short* h    = (unsigned short*)take((size_t)EEXP * CCAP * FDIM * 2);
    float*          eo   = (float*)         take((size_t)EEXP * CCAP * MDIM * 4);
    (void)ws_size; (void)in_sizes; (void)n_in; (void)out_size;

    gate_kernel<<<S_TOK / 8, 256, 0, stream>>>(x, wg, idx, prob);
    scan_kernel<<<1, 1024, 0, stream>>>(idx, loc);
    dispatch_kernel<<<S_TOK, 256, 0, stream>>>(x, idx, loc, disp);

    // FFN1: [C,M] x [M,F] per expert, fused GELU, bf16 out
    moe_gemm_kernel<true><<<dim3(FDIM / 128, CCAP / 128, EEXP), 256, 0, stream>>>(
        disp, w1, b1, h, nullptr, CCAP, MDIM, FDIM);
    // FFN2: [C,F] x [F,M] per expert, fp32 out
    moe_gemm_kernel<false><<<dim3(MDIM / 128, CCAP / 128, EEXP), 256, 0, stream>>>(
        h, w2, b2, nullptr, eo, CCAP, FDIM, MDIM);

    combine_kernel<<<S_TOK, 256, 0, stream>>>(eo, idx, loc, prob, out);
}
